// MonLipLayer_23931557773622
// MI455X (gfx1250) — matrix operand order repaired and checked by reading, not compiled, not run on hardware
//
#include <hip/hip_runtime.h>
#include <hip/hip_bf16.h>
#include <math.h>

// ---------------------------------------------------------------------------
// MonLipLayer on MI455X (gfx1250, wave32, WMMA).
// Heavy GEMMs in bf16 WMMA (v_wmma_f32_16x16x32_bf16, f32 accumulate).
// Cayley setup (norms, skew+Gram assembly, Gauss-Jordan inverse) in fp32.
// ---------------------------------------------------------------------------

#define EPS_  1e-5f
#define MUNU_ 10.1f   // MU + NU

typedef __attribute__((ext_vector_type(16))) __bf16 v16bf;
typedef __attribute__((ext_vector_type(8)))  __bf16 v8bf;
typedef __attribute__((ext_vector_type(8)))  float  v8f;

__device__ __forceinline__ __bf16 f2bf(float f) {
  unsigned u = __builtin_bit_cast(unsigned, f);
  u += 0x7fffu + ((u >> 16) & 1u);               // round-to-nearest-even
  unsigned short h = (unsigned short)(u >> 16);
  return __builtin_bit_cast(__bf16, h);
}
__device__ __forceinline__ float bf2f(__bf16 b) {
  unsigned short h = __builtin_bit_cast(unsigned short, b);
  unsigned u = ((unsigned)h) << 16;
  return __builtin_bit_cast(float, u);
}

// ---------------------------------------------------------------------------
// Setup kernels (fp32)
// ---------------------------------------------------------------------------

__global__ void zero_f32(float* p, int n) {
  int i = blockIdx.x * blockDim.x + threadIdx.x;
  if (i < n) p[i] = 0.f;
}

__global__ void sumsq_kernel(const float* __restrict__ p, long long n, float* acc) {
  __shared__ float red[256];
  float s = 0.f;
  for (long long i = (long long)blockIdx.x * blockDim.x + threadIdx.x; i < n;
       i += (long long)gridDim.x * blockDim.x) {
    float v = p[i]; s += v * v;
  }
  red[threadIdx.x] = s; __syncthreads();
  for (int st = 128; st > 0; st >>= 1) {
    if (threadIdx.x < st) red[threadIdx.x] += red[threadIdx.x + st];
    __syncthreads();
  }
  if (threadIdx.x == 0) atomicAdd(acc, red[0]);
}

// M = I + A, N = I - A, B2 = I   for Q  (n=1024, A = s(U-U^T) + s^2 V^T V)
__global__ void build_Mq(const float* __restrict__ Fq, const float* __restrict__ fq,
                         const float* __restrict__ ssq,
                         float* __restrict__ M, float* __restrict__ Nn, float* __restrict__ B2) {
  int j = blockIdx.x * 16 + threadIdx.x;
  int i = blockIdx.y * 16 + threadIdx.y;
  float s = fq[0] / (sqrtf(ssq[0]) + EPS_);
  float skew = s * (Fq[(size_t)i * 1024 + j] - Fq[(size_t)j * 1024 + i]);
  float dot = 0.f;
  for (int k = 0; k < 3072; ++k)
    dot += Fq[(size_t)(1024 + k) * 1024 + i] * Fq[(size_t)(1024 + k) * 1024 + j];
  float a = skew + s * s * dot;
  float d = (i == j) ? 1.f : 0.f;
  M [(size_t)i * 1024 + j] = d + a;
  Nn[(size_t)i * 1024 + j] = d - a;
  B2[(size_t)i * 1024 + j] = d;
}

// Same for R_k (n=512).  k==0: A = s(Fr0 - Fr0^T).
// k>=1 (on W = (s*Fr)^T): A[i][j] = s(Fr[j][i]-Fr[i][j]) + s^2 sum_m Fr[i][512+m]Fr[j][512+m]
__global__ void build_Mr(const float* __restrict__ Fr0, const float* __restrict__ FrR,
                         const float* __restrict__ fr, const float* __restrict__ ssq, int k,
                         float* __restrict__ M, float* __restrict__ Nn, float* __restrict__ B2) {
  int j = blockIdx.x * 16 + threadIdx.x;
  int i = blockIdx.y * 16 + threadIdx.y;
  float a;
  if (k == 0) {
    float s = fr[0] / (sqrtf(ssq[1]) + EPS_);
    a = s * (Fr0[(size_t)i * 512 + j] - Fr0[(size_t)j * 512 + i]);
  } else {
    const float* F = FrR + (size_t)(k - 1) * 512 * 1024;
    float s = fr[k] / (sqrtf(ssq[1 + k]) + EPS_);
    float skew = s * (F[(size_t)j * 1024 + i] - F[(size_t)i * 1024 + j]);
    float dot = 0.f;
    for (int m = 0; m < 512; ++m)
      dot += F[(size_t)i * 1024 + 512 + m] * F[(size_t)j * 1024 + 512 + m];
    a = skew + s * s * dot;
  }
  float d = (i == j) ? 1.f : 0.f;
  M [(size_t)i * 512 + j] = d + a;
  Nn[(size_t)i * 512 + j] = d - a;
  B2[(size_t)i * 512 + j] = d;
}

// Gauss-Jordan on [M | B1 | B2], no pivoting (I+A has PD symmetric part).
// Single workgroup; after completion B1 = inv(M)*B1_in, B2 = inv(M)*B2_in.
__global__ void __launch_bounds__(1024)
gj_kernel(float* __restrict__ M, float* __restrict__ B1, float* __restrict__ B2, int n) {
  __shared__ float pivinv;
  for (int col = 0; col < n; ++col) {
    if (threadIdx.x == 0) pivinv = 1.f / M[(size_t)col * n + col];
    __syncthreads();
    float pv = pivinv;
    for (int c = threadIdx.x; c < n; c += blockDim.x) {
      M [(size_t)col * n + c] *= pv;
      B1[(size_t)col * n + c] *= pv;
      B2[(size_t)col * n + c] *= pv;
    }
    __syncthreads();
    for (int r = threadIdx.x; r < n; r += blockDim.x) {
      if (r == col) continue;
      float f = M[(size_t)r * n + col];
      if (f != 0.f) {
        for (int c = 0; c < n; ++c) {
          M [(size_t)r * n + c] -= f * M [(size_t)col * n + c];
          B1[(size_t)r * n + c] -= f * B1[(size_t)col * n + c];
          B2[(size_t)r * n + c] -= f * B2[(size_t)col * n + c];
        }
      }
    }
    __syncthreads();
  }
}

// Q top rows: Q[i][j] = (inv(M)N)[i][j], emitted row-major + transposed, bf16.
__global__ void emit_Q_top(const float* __restrict__ Qtop, __bf16* Qbf, __bf16* QT) {
  int j = blockIdx.x * 16 + threadIdx.x;
  int i = blockIdx.y * 16 + threadIdx.y;
  float q = Qtop[(size_t)i * 1024 + j];
  Qbf[(size_t)i * 1024 + j] = f2bf(q);
  QT [(size_t)j * 4096 + i] = f2bf(q);
}

// Q bottom rows: Q[1024+v][j] = -2 s (Fq_bot @ inv(M))[v][j]
__global__ void emit_Q_bot(const float* __restrict__ Fq, const float* __restrict__ fq,
                           const float* __restrict__ ssq, const float* __restrict__ Inv,
                           __bf16* Qbf, __bf16* QT) {
  int j = blockIdx.x * 16 + threadIdx.x;
  int v = blockIdx.y * 16 + threadIdx.y;   // < 3072
  float s = fq[0] / (sqrtf(ssq[0]) + EPS_);
  float dot = 0.f;
  for (int i = 0; i < 1024; ++i)
    dot += Fq[(size_t)(1024 + v) * 1024 + i] * Inv[(size_t)i * 1024 + j];
  float q = -2.f * s * dot;
  Qbf[(size_t)(1024 + v) * 1024 + j] = f2bf(q);
  QT [(size_t)j * 4096 + 1024 + v]   = f2bf(q);
}

// R left 512 cols.  k==0: R = G1 directly.  k>=1: R[r][c] = G1[c][r].
__global__ void emit_R_left(const float* __restrict__ G1, __bf16* R, __bf16* RT,
                            int ldR, int transposed) {
  int c = blockIdx.x * 16 + threadIdx.x;
  int r = blockIdx.y * 16 + threadIdx.y;
  float q = transposed ? G1[(size_t)c * 512 + r] : G1[(size_t)r * 512 + c];
  R [(size_t)r * ldR + c] = f2bf(q);
  RT[(size_t)c * 512 + r] = f2bf(q);
}

// R right 512 cols (k>=1): R[r][512+c2] = -2 s sum_i Fr[i][512+c2] * Inv[i][r]
__global__ void emit_R_right(const float* __restrict__ F, const float* __restrict__ fr,
                             const float* __restrict__ ssq, int k,
                             const float* __restrict__ Inv, __bf16* R, __bf16* RT) {
  int r  = blockIdx.x * 16 + threadIdx.x;
  int c2 = blockIdx.y * 16 + threadIdx.y;
  float s = fr[k] / (sqrtf(ssq[1 + k]) + EPS_);
  float dot = 0.f;
  for (int i = 0; i < 512; ++i)
    dot += F[(size_t)i * 1024 + 512 + c2] * Inv[(size_t)i * 512 + r];
  float q = -2.f * s * dot;
  R [(size_t)r * 1024 + 512 + c2] = f2bf(q);
  RT[(size_t)(512 + c2) * 512 + r] = f2bf(q);
}

__global__ void f32_to_bf16(const float* __restrict__ src, __bf16* __restrict__ dst,
                            long long n) {
  for (long long i = (long long)blockIdx.x * blockDim.x + threadIdx.x; i < n;
       i += (long long)gridDim.x * blockDim.x)
    dst[i] = f2bf(src[i]);
}

// YH[:, 3584:4096] = H (final hk_1)
__global__ void copy_h_to_yh(const __bf16* __restrict__ H, __bf16* __restrict__ YH) {
  long long idx = (long long)blockIdx.x * blockDim.x + threadIdx.x;
  int m = (int)(idx >> 9), c = (int)(idx & 511);
  YH[(size_t)m * 4096 + 3584 + c] = H[idx];
}

// ---------------------------------------------------------------------------
// Generic bf16 WMMA GEMM.  C[m][n] = alpha * sum_k A[m][k] * BT[n][k]  (+ epilogue)
// A is split in K at K0 between pointers A0/A1 (handles concat([xk, hk_1])).
// Block: 256 threads = 8 waves (2x4), block tile 64(M) x 256(N),
// wave tile 32x64 = 2x4 tiles of 16x16, K-step 32.
// ---------------------------------------------------------------------------

#define MODE_STORE     0   // store bf16 OUT
#define MODE_BIAS_RELU 1   // relu(acc + bias[n]) -> bf16 OUT
#define MODE_SUB       2   // acc - sub[m][n] -> bf16 OUT
#define MODE_LAYER     3   // n<512: acc - sub -> OUT ; n>=512: hprev - acc -> out2
#define MODE_FINAL     4   // 0.5*(MUNU*x + acc) + by -> f32 OUT

__global__ void __launch_bounds__(256) wmma_gemm_kernel(
    const __bf16* __restrict__ A0, int lda0, int K0,
    const __bf16* __restrict__ A1, int lda1,
    const __bf16* __restrict__ BT, int ldbt,
    int Ktot, float alpha, int mode,
    const float*  __restrict__ bias,
    const __bf16* __restrict__ sub,   int ld_sub,
    const __bf16* __restrict__ hprev, int ld_hprev,
    __bf16*       __restrict__ out2,  int ld_out2,
    const float*  __restrict__ xin,   int ld_x,
    const float*  __restrict__ byv,
    void*         __restrict__ OUT,   int ldo)
{
  const int tid  = threadIdx.x;
  const int lane = tid & 31;
  const int w    = tid >> 5;
  const int wr   = w >> 2, wc = w & 3;
  const int m0   = blockIdx.y * 64  + wr * 32;
  const int n0   = blockIdx.x * 256 + wc * 64;
  const int l15  = lane & 15;       // A: M-row within tile; B: N-col; D: N-col
  const int kh   = lane >> 4;

  v8f zero = {0.f, 0.f, 0.f, 0.f, 0.f, 0.f, 0.f, 0.f};
  v8f acc[2][4];
#pragma unroll
  for (int mt = 0; mt < 2; ++mt)
#pragma unroll
    for (int nt = 0; nt < 4; ++nt) acc[mt][nt] = zero;

  for (int kb = 0; kb < Ktot; kb += 32) {
    const __bf16* Ap; int lda, kloc;
    if (kb < K0) { Ap = A0; lda = lda0; kloc = kb; }
    else         { Ap = A1; lda = lda1; kloc = kb - K0; }

    // A fragments: 16x32 bf16.  lane = M-row (l15); 16 contiguous K from 16*kh
    v16bf af[2];
#pragma unroll
    for (int mt = 0; mt < 2; ++mt) {
      const __bf16* base = Ap + (size_t)(m0 + mt * 16 + l15) * lda + kloc;
      v8bf c0 = *(const v8bf*)(base + 16 * kh);
      v8bf c1 = *(const v8bf*)(base + 16 * kh + 8);
#pragma unroll
      for (int e = 0; e < 8; ++e) { af[mt][e] = c0[e]; af[mt][8 + e] = c1[e]; }
    }

    // B fragments: 32x16 bf16 from transposed weights BT[n][k]; lane = column,
    // holds 16 contiguous K values starting at kb + 16*kh.
    v16bf bfv[4];
#pragma unroll
    for (int nt = 0; nt < 4; ++nt) {
      const __bf16* bp = BT + (size_t)(n0 + nt * 16 + l15) * ldbt + kb + 16 * kh;
      bfv[nt] = *(const v16bf*)bp;
      if (kb + 32 < Ktot) __builtin_prefetch(bp + 32, 0, 1);   // global_prefetch
    }

#pragma unroll
    for (int mt = 0; mt < 2; ++mt)
#pragma unroll
      for (int nt = 0; nt < 4; ++nt)
        acc[mt][nt] = __builtin_amdgcn_wmma_f32_16x16x32_bf16(
            false, af[mt], false, bfv[nt], (short)0, acc[mt][nt], false, false);
  }

  // Epilogue.  D layout: element v of lane -> m = v + 8*kh, n = lane&15.
#pragma unroll
  for (int mt = 0; mt < 2; ++mt) {
#pragma unroll
    for (int nt = 0; nt < 4; ++nt) {
      const int nIdx  = n0 + nt * 16 + l15;
      const int mBase = m0 + mt * 16 + 8 * kh;
#pragma unroll
      for (int v = 0; v < 8; ++v) {
        const int m = mBase + v;
        float a = alpha * acc[mt][nt][v];
        if (mode == MODE_STORE) {
          ((__bf16*)OUT)[(size_t)m * ldo + nIdx] = f2bf(a);
        } else if (mode == MODE_BIAS_RELU) {
          float t = a + bias[nIdx];
          ((__bf16*)OUT)[(size_t)m * ldo + nIdx] = f2bf(t > 0.f ? t : 0.f);
        } else if (mode == MODE_SUB) {
          float t = a - bf2f(sub[(size_t)m * ld_sub + nIdx]);
          ((__bf16*)OUT)[(size_t)m * ldo + nIdx] = f2bf(t);
        } else if (mode == MODE_LAYER) {
          if (nIdx < 512) {
            float t = a - bf2f(sub[(size_t)m * ld_sub + nIdx]);
            ((__bf16*)OUT)[(size_t)m * ldo + nIdx] = f2bf(t);
          } else {
            float t = bf2f(hprev[(size_t)m * ld_hprev + (nIdx - 512)]) - a;
            out2[(size_t)m * ld_out2 + (nIdx - 512)] = f2bf(t);
          }
        } else {  // MODE_FINAL
          float t = 0.5f * (MUNU_ * xin[(size_t)m * ld_x + nIdx] + a) + byv[nIdx];
          ((float*)OUT)[(size_t)m * ldo + nIdx] = t;
        }
      }
    }
  }
}

// ---------------------------------------------------------------------------
// Host orchestration
// ---------------------------------------------------------------------------

static inline size_t align256(size_t x) { return (x + 255) & ~(size_t)255; }

extern "C" void kernel_launch(void* const* d_in, const int* in_sizes, int n_in,
                              void* d_out, int out_size, void* d_ws, size_t ws_size,
                              hipStream_t stream) {
  (void)in_sizes; (void)n_in; (void)out_size; (void)ws_size;
  const float* x   = (const float*)d_in[0];   // 8192 x 1024
  const float* Fq  = (const float*)d_in[1];   // 4096 x 1024
  const float* fq  = (const float*)d_in[2];   // 1
  const float* by  = (const float*)d_in[3];   // 1024
  const float* Fr0 = (const float*)d_in[4];   // 512 x 512
  const float* FrR = (const float*)d_in[5];   // 7 x 512 x 1024
  const float* fr  = (const float*)d_in[6];   // 8
  const float* bb  = (const float*)d_in[7];   // 8 x 512
  float* out = (float*)d_out;

  char* ws = (char*)d_ws;
  size_t off = 0;
  auto alloc = [&](size_t bytes) -> char* { char* p = ws + off; off = align256(off + bytes); return p; };

  float* ssq = (float*)alloc(16 * sizeof(float));
  float* Mq  = (float*)alloc((size_t)1024 * 1024 * 4);
  float* Nq  = (float*)alloc((size_t)1024 * 1024 * 4);
  float* B2q = (float*)alloc((size_t)1024 * 1024 * 4);
  float* Mr  = (float*)alloc((size_t)512 * 512 * 4);
  float* Nr  = (float*)alloc((size_t)512 * 512 * 4);
  float* B2r = (float*)alloc((size_t)512 * 512 * 4);
  __bf16* Qbf  = (__bf16*)alloc((size_t)4096 * 1024 * 2);
  __bf16* QTbf = (__bf16*)alloc((size_t)1024 * 4096 * 2);
  __bf16* Rbf  = (__bf16*)alloc((size_t)8 * 512 * 1024 * 2);
  __bf16* RTbf = (__bf16*)alloc((size_t)8 * 1024 * 512 * 2);
  __bf16* xbf  = (__bf16*)alloc((size_t)8192 * 1024 * 2);
  __bf16* xh   = (__bf16*)alloc((size_t)8192 * 4096 * 2);
  __bf16* P    = (__bf16*)alloc((size_t)8192 * 512 * 2);
  __bf16* H0   = (__bf16*)alloc((size_t)8192 * 512 * 2);
  __bf16* H1   = (__bf16*)alloc((size_t)8192 * 512 * 2);
  __bf16* YH   = (__bf16*)alloc((size_t)8192 * 4096 * 2);

  const float SQRT_GAM = sqrtf(10.0f - 0.1f);
  const float SQRT2    = sqrtf(2.0f);
  dim3 blk16(16, 16);

  // ---- norms ----
  zero_f32<<<1, 64, 0, stream>>>(ssq, 16);
  sumsq_kernel<<<512, 256, 0, stream>>>(Fq, (long long)4096 * 1024, ssq + 0);
  sumsq_kernel<<<256, 256, 0, stream>>>(Fr0, (long long)512 * 512, ssq + 1);
  for (int k = 1; k < 8; ++k)
    sumsq_kernel<<<256, 256, 0, stream>>>(FrR + (size_t)(k - 1) * 512 * 1024,
                                          (long long)512 * 1024, ssq + 1 + k);

  // ---- Cayley for Q ----
  build_Mq<<<dim3(64, 64), blk16, 0, stream>>>(Fq, fq, ssq, Mq, Nq, B2q);
  gj_kernel<<<1, 1024, 0, stream>>>(Mq, Nq, B2q, 1024);
  emit_Q_top<<<dim3(64, 64), blk16, 0, stream>>>(Nq, Qbf, QTbf);
  emit_Q_bot<<<dim3(64, 192), blk16, 0, stream>>>(Fq, fq, ssq, B2q, Qbf, QTbf);

  // ---- Cayley for R_k ----
  for (int k = 0; k < 8; ++k) {
    build_Mr<<<dim3(32, 32), blk16, 0, stream>>>(Fr0, FrR, fr, ssq, k, Mr, Nr, B2r);
    gj_kernel<<<1, 512, 0, stream>>>(Mr, Nr, B2r, 512);
    __bf16* Rk  = Rbf  + (size_t)k * 512 * 1024;
    __bf16* RTk = RTbf + (size_t)k * 1024 * 512;
    emit_R_left<<<dim3(32, 32), blk16, 0, stream>>>(Nr, Rk, RTk, (k == 0 ? 512 : 1024), (k != 0));
    if (k >= 1)
      emit_R_right<<<dim3(32, 32), blk16, 0, stream>>>(
          FrR + (size_t)(k - 1) * 512 * 1024, fr, ssq, k, B2r, Rk, RTk);
  }

  // ---- main pipeline ----
  f32_to_bf16<<<2048, 256, 0, stream>>>(x, xbf, (long long)8192 * 1024);

  // xh = sqrt_gam * x @ Q^T   (BT = Q row-major [u][c])
  wmma_gemm_kernel<<<dim3(16, 128), 256, 0, stream>>>(
      xbf, 1024, 1024, nullptr, 0, Qbf, 1024, 1024, SQRT_GAM, MODE_STORE,
      nullptr, nullptr, 0, nullptr, 0, nullptr, 0, nullptr, 0, nullptr, xh, 4096);

  // layer 0: P = relu(sqrt2 * xh[:, :512] @ R0^T + b0);  H0 = sqrt2 * P @ R0 - xh[:, :512]
  wmma_gemm_kernel<<<dim3(2, 128), 256, 0, stream>>>(
      xh, 4096, 512, nullptr, 0, Rbf, 512, 512, SQRT2, MODE_BIAS_RELU,
      bb, nullptr, 0, nullptr, 0, nullptr, 0, nullptr, 0, nullptr, P, 512);
  wmma_gemm_kernel<<<dim3(2, 128), 256, 0, stream>>>(
      P, 512, 512, nullptr, 0, RTbf, 512, 512, SQRT2, MODE_SUB,
      nullptr, xh, 4096, nullptr, 0, nullptr, 0, nullptr, 0, nullptr, H0, 512);

  __bf16* Hp = H0; __bf16* Hc = H1;
  for (int k = 1; k < 8; ++k) {
    const __bf16* Rk  = Rbf  + (size_t)k * 512 * 1024;
    const __bf16* RTk = RTbf + (size_t)k * 1024 * 512;
    // P = relu(sqrt2 * [xk, h_{k-1}] @ Rk^T + b_k)
    wmma_gemm_kernel<<<dim3(2, 128), 256, 0, stream>>>(
        xh + (size_t)k * 512, 4096, 512, Hp, 512, Rk, 1024, 1024, SQRT2, MODE_BIAS_RELU,
        bb + (size_t)k * 512, nullptr, 0, nullptr, 0, nullptr, 0, nullptr, 0, nullptr, P, 512);
    // gh = sqrt2 * P @ Rk ;  Hc = gh[:, :512] - xk ;  YH part = h_{k-1} - gh[:, 512:]
    wmma_gemm_kernel<<<dim3(4, 128), 256, 0, stream>>>(
        P, 512, 512, nullptr, 0, RTk, 512, 512, SQRT2, MODE_LAYER,
        nullptr, xh + (size_t)k * 512, 4096, Hp, 512,
        YH + (size_t)(k - 1) * 512, 4096, nullptr, 0, nullptr, Hc, 512);
    __bf16* t = Hp; Hp = Hc; Hc = t;
  }

  // YH[:, 3584:] = h7
  copy_h_to_yh<<<(8192 * 512) / 256, 256, 0, stream>>>(Hp, YH);

  // out = 0.5*((mu+nu)*x + sqrt_gam * YH @ Q) + by   (BT = Q^T [c][u])
  wmma_gemm_kernel<<<dim3(4, 128), 256, 0, stream>>>(
      YH, 4096, 4096, nullptr, 0, QTbf, 4096, 4096, SQRT_GAM, MODE_FINAL,
      nullptr, nullptr, 0, nullptr, 0, nullptr, 0, x, 1024, by, out, 1024);
}
